// SelfAttention_3779571220959
// MI455X (gfx1250) — matrix operand order repaired and checked by reading, not compiled, not run on hardware
//
#include <hip/hip_runtime.h>
#include <hip/hip_bf16.h>
#include <stdint.h>

typedef _Float16 f16;
typedef _Float16 half16 __attribute__((ext_vector_type(16)));
typedef float    floatx8 __attribute__((ext_vector_type(8)));

#define C_DIM 1024
#define P_DIM 2048
#define B_DIM 4
#define NHEADS 16
#define HEAD 64
#define M_DIM (B_DIM * P_DIM) /* 8192 */

__device__ __forceinline__ floatx8 wmma_f16f32(half16 a, half16 b, floatx8 c) {
  return __builtin_amdgcn_wmma_f32_16x16x32_f16(
      /*neg_a=*/false, a, /*neg_b=*/false, b,
      /*c_mod=*/(short)0, c, /*reuse_a=*/false, /*reuse_b=*/false);
}

union FragU { half16 v; uint4 q[2]; };

// A-frag (16x32 f16): lanes 0-15 hold K 0..7 & 16..23 of row M=lane; lanes 16-31 hold K 8..15 & 24..31.
__device__ __forceinline__ half16 load_frag_a(const f16* base, int ld) {
  const int lane = threadIdx.x & 31;
  const f16* p = base + (lane & 15) * ld + (lane >> 4) * 8;
  FragU f;
  f.q[0] = *(const uint4*)(p);
  f.q[1] = *(const uint4*)(p + 16);
  return f.v;
}

// B-frag (32x16 f16) from an N-major [n][k] buffer: column n=lane&15;
// lanes 0-15 hold K 0..7 & 16..23, lanes 16-31 hold K 8..15 & 24..31.
__device__ __forceinline__ half16 load_frag_b(const f16* base, int ld) {
  const int lane = threadIdx.x & 31;
  const f16* p = base + (lane & 15) * ld + (lane >> 4) * 8;
  FragU f;
  f.q[0] = *(const uint4*)(p);
  f.q[1] = *(const uint4*)(p + 16);
  return f.v;
}

// ---- CDNA5 async global->LDS copy (ASYNCcnt-tracked DMA-style load, GV addressing) ----
// LDS byte address is taken per lane from the VDST VGPR; generic shared pointers are {aperture_hi, lds_off32},
// so truncating the flat pointer yields the per-lane LDS byte offset.
__device__ __forceinline__ void async_copy_b128(f16* lds_ptr, const f16* gptr) {
  unsigned lds_off = (unsigned)(uintptr_t)lds_ptr;
  asm volatile("global_load_async_to_lds_b128 %0, %1, off"
               :: "v"(lds_off), "v"(gptr)
               : "memory");
}
__device__ __forceinline__ void wait_async_le4() {
  asm volatile("s_wait_asynccnt 0x4" ::: "memory");
}
__device__ __forceinline__ void wait_async_le8() {
  asm volatile("s_wait_asynccnt 0x8" ::: "memory");
}
__device__ __forceinline__ void wait_async_0() {
  asm volatile("s_wait_asynccnt 0x0" ::: "memory");
}

// ---------------- conversion kernels ----------------
__global__ void cvt_f32_f16(const float* __restrict__ src, f16* __restrict__ dst, int n) {
  int i = (blockIdx.x * blockDim.x + threadIdx.x) * 4;
  if (i < n) {
    float4 v = *(const float4*)(src + i);
    dst[i + 0] = (f16)v.x; dst[i + 1] = (f16)v.y;
    dst[i + 2] = (f16)v.z; dst[i + 3] = (f16)v.w;
  }
}

// W is [K=1024][N=1024] fp32 row-major -> Wt [N][K] fp16 row-major
__global__ void cvtT_f32_f16(const float* __restrict__ W, f16* __restrict__ Wt) {
  int i = blockIdx.x * blockDim.x + threadIdx.x;
  int k = i >> 10, n = i & 1023;
  Wt[(size_t)n * C_DIM + k] = (f16)W[i];
}

// ---------------- tiled WMMA GEMM: C[M,N] = A[M,K] @ Bt[N,K]^T + bias ----------------
// MODE 0: fp32 row-major [M,C] (final output)
// MODE 1: fp16 head-major [B,H,P,64]   (Q, K)
// MODE 2: fp16 head-major transposed [B,H,64,P]  (V)
template <int MODE>
__global__ __launch_bounds__(256)
void gemm128(const f16* __restrict__ A, const f16* __restrict__ Bt,
             const float* __restrict__ bias, float* __restrict__ outf,
             f16* __restrict__ outh) {
  constexpr int KT  = 32;
  constexpr int LDT = 40; // 80B rows: 16B-aligned for b128, bank-spread
  __shared__ f16 As[2][128 * LDT];
  __shared__ f16 Bs[2][128 * LDT];

  const int tid = threadIdx.x;
  const int lane = tid & 31;
  const int wave = tid >> 5;
  const int wm = wave >> 1; // 0..3 : 32-row strip
  const int wn = wave & 1;  // 0..1 : 64-col strip
  const int m0 = blockIdx.y * 128;
  const int n0 = blockIdx.x * 128;

  const f16* Ag = A + (size_t)m0 * C_DIM;
  const f16* Bg = Bt + (size_t)n0 * C_DIM;

  floatx8 acc[2][4] = {};

  // async tile stage: 128 rows x 32 halves = 512 b128s; 2 per thread per matrix
  auto stage = [&](const f16* gA, const f16* gB, f16* sA, f16* sB) {
#pragma unroll
    for (int i = 0; i < 2; i++) {
      int idx = tid + i * 256;               // 0..511
      int row = idx >> 2, cb = (idx & 3) * 8;
      async_copy_b128(&sA[row * LDT + cb], &gA[(size_t)row * C_DIM + cb]);
      async_copy_b128(&sB[row * LDT + cb], &gB[(size_t)row * C_DIM + cb]);
    }
  };

  stage(Ag, Bg, As[0], Bs[0]);

  for (int kt = 0; kt < C_DIM; kt += KT) {
    const int buf = (kt >> 5) & 1;
    if (kt + KT < C_DIM) {
      stage(Ag + kt + KT, Bg + kt + KT, As[buf ^ 1], Bs[buf ^ 1]);
      wait_async_le4();   // current tile's 4 loads complete; next tile in flight
    } else {
      wait_async_0();
    }
    __syncthreads();

    half16 a0 = load_frag_a(&As[buf][(wm * 32 + 0) * LDT], LDT);
    half16 a1 = load_frag_a(&As[buf][(wm * 32 + 16) * LDT], LDT);
#pragma unroll
    for (int j = 0; j < 4; j++) {
      half16 b = load_frag_b(&Bs[buf][(wn * 64 + j * 16) * LDT], LDT);
      acc[0][j] = wmma_f16f32(a0, b, acc[0][j]);
      acc[1][j] = wmma_f16f32(a1, b, acc[1][j]);
    }
    __syncthreads();
  }

  const int cn = lane & 15, hi = lane >> 4;
#pragma unroll
  for (int i = 0; i < 2; i++)
#pragma unroll
    for (int j = 0; j < 4; j++) {
      int nn = n0 + wn * 64 + j * 16 + cn;
      float bv = bias[nn];
#pragma unroll
      for (int r = 0; r < 8; r++) {
        int mm = m0 + wm * 32 + i * 16 + r + 8 * hi;
        float val = acc[i][j][r] + bv;
        if (MODE == 0) {
          outf[(size_t)mm * C_DIM + nn] = val;
        } else {
          int bb = mm >> 11, pp = mm & (P_DIM - 1);
          int hh = nn >> 6, dd = nn & 63;
          if (MODE == 1)
            outh[((size_t)(bb * NHEADS + hh) * P_DIM + pp) * HEAD + dd] = (f16)val;
          else // MODE 2: transposed per head [B,H,64,P]; lane's 8 r-values hit contiguous p
            outh[((size_t)(bb * NHEADS + hh) * HEAD + dd) * P_DIM + pp] = (f16)val;
        }
      }
    }
}

// ---------------- flash attention: 64-query tile per block per (b,h) ----------------
// Q,K head-major [B,H,P,64]; V pre-transposed [B,H,64,P] so both K and V tiles are
// contiguous copies -> async double-buffered.
__global__ __launch_bounds__(128)
void attn64(const f16* __restrict__ Qh, const f16* __restrict__ Kh,
            const f16* __restrict__ Vt, f16* __restrict__ Yt) {
  constexpr int LD = 80; // 160B rows: 16B-aligned
  __shared__ f16 Ks[2][64 * LD];  // [key][d]
  __shared__ f16 Vs[2][64 * LD];  // [d][key]
  __shared__ f16 Ps[4][16 * LD];  // per-wave P staging [row][key]

  const int tid = threadIdx.x;
  const int lane = tid & 31;
  const int wave = tid >> 5;
  const int cn = lane & 15, hi = lane >> 4;

  const int bh = blockIdx.y; // b*16 + h
  const int q0 = blockIdx.x * 64;
  const f16* Qp  = Qh + (size_t)bh * P_DIM * HEAD;
  const f16* Kp  = Kh + (size_t)bh * P_DIM * HEAD;
  const f16* Vtp = Vt + (size_t)bh * HEAD * P_DIM;

  // Q fragments for this wave's 16 query rows, register-resident across the loop
  const f16* qbase = Qp + (size_t)(q0 + wave * 16) * HEAD;
  half16 qa0 = load_frag_a(qbase, HEAD);       // d 0..31
  half16 qa1 = load_frag_a(qbase + 32, HEAD);  // d 32..63

  floatx8 accy[4] = {};
  float rmax[8], rsum[8];
#pragma unroll
  for (int r = 0; r < 8; r++) { rmax[r] = -1e30f; rsum[r] = 0.0f; }

  const float scale = 0.125f; // HEAD^-0.5

  // 64 rows x 64 halves per tile = 512 b128s; 4 per thread per matrix
  auto stage = [&](int j0, f16* sK, f16* sV) {
#pragma unroll
    for (int i = 0; i < 4; i++) {
      int idx = tid + i * 128;               // 0..511
      int row = idx >> 3, cb = (idx & 7) * 8;
      async_copy_b128(&sK[row * LD + cb], &Kp[(size_t)(j0 + row) * HEAD + cb]);
      async_copy_b128(&sV[row * LD + cb], &Vtp[(size_t)row * P_DIM + j0 + cb]);
    }
  };

  stage(0, Ks[0], Vs[0]);

  for (int j0 = 0; j0 < P_DIM; j0 += 64) {
    const int buf = (j0 >> 6) & 1;
    if (j0 + 64 < P_DIM) {
      stage(j0 + 64, Ks[buf ^ 1], Vs[buf ^ 1]);
      wait_async_le8();
    } else {
      wait_async_0();
    }
    __syncthreads();

    // S = scale * Q K^T  (16 q-rows x 64 keys per wave)
    floatx8 s[4];
#pragma unroll
    for (int kg = 0; kg < 4; kg++) {
      half16 b0 = load_frag_b(&Ks[buf][(kg * 16) * LD], LD);
      half16 b1 = load_frag_b(&Ks[buf][(kg * 16) * LD + 32], LD);
      floatx8 z = {};
      floatx8 t0 = wmma_f16f32(qa0, b0, z);
      s[kg] = wmma_f16f32(qa1, b1, t0);
      s[kg] = s[kg] * scale;
    }

    // online softmax update (row = r + 8*hi, 16 lanes per row-half)
#pragma unroll
    for (int r = 0; r < 8; r++) {
      float tmax = -1e30f;
#pragma unroll
      for (int kg = 0; kg < 4; kg++) tmax = fmaxf(tmax, s[kg][r]);
      tmax = fmaxf(tmax, __shfl_xor(tmax, 1));
      tmax = fmaxf(tmax, __shfl_xor(tmax, 2));
      tmax = fmaxf(tmax, __shfl_xor(tmax, 4));
      tmax = fmaxf(tmax, __shfl_xor(tmax, 8));
      float mnew = fmaxf(rmax[r], tmax);
      float corr = __expf(rmax[r] - mnew);
      rmax[r] = mnew;
      float rs = 0.0f;
#pragma unroll
      for (int kg = 0; kg < 4; kg++) {
        float pv = __expf(s[kg][r] - mnew);
        s[kg][r] = pv;
        rs += pv;
      }
      rs += __shfl_xor(rs, 1);
      rs += __shfl_xor(rs, 2);
      rs += __shfl_xor(rs, 4);
      rs += __shfl_xor(rs, 8);
      rsum[r] = rsum[r] * corr + rs;
#pragma unroll
      for (int dg = 0; dg < 4; dg++) accy[dg][r] = accy[dg][r] * corr;
    }

    // stage P (fp16) through per-wave LDS to reshape C-layout -> A-frag layout
    f16* pbase = &Ps[wave][0];
#pragma unroll
    for (int kg = 0; kg < 4; kg++)
#pragma unroll
      for (int r = 0; r < 8; r++)
        pbase[(r + 8 * hi) * LD + kg * 16 + cn] = (f16)s[kg][r];

    half16 pa0 = load_frag_a(pbase, LD);       // keys 0..31
    half16 pa1 = load_frag_a(pbase + 32, LD);  // keys 32..63

    // Y += P @ V
#pragma unroll
    for (int dg = 0; dg < 4; dg++) {
      half16 vb0 = load_frag_b(&Vs[buf][(dg * 16) * LD], LD);
      half16 vb1 = load_frag_b(&Vs[buf][(dg * 16) * LD + 32], LD);
      accy[dg] = wmma_f16f32(pa1, vb1, wmma_f16f32(pa0, vb0, accy[dg]));
    }
    __syncthreads();
  }

  // normalize and write token-major fp16 [B*P, C]
  const int b = bh >> 4, h = bh & 15;
#pragma unroll
  for (int dg = 0; dg < 4; dg++)
#pragma unroll
    for (int r = 0; r < 8; r++) {
      int p = q0 + wave * 16 + r + 8 * hi;
      float val = accy[dg][r] / rsum[r];
      int nchan = h * 64 + dg * 16 + cn;
      Yt[(size_t)(b * P_DIM + p) * C_DIM + nchan] = (f16)val;
    }
}

extern "C" void kernel_launch(void* const* d_in, const int* in_sizes, int n_in,
                              void* d_out, int out_size, void* d_ws, size_t ws_size,
                              hipStream_t stream) {
  const float* x  = (const float*)d_in[0];
  const float* Wq = (const float*)d_in[1];
  const float* bq = (const float*)d_in[2];
  const float* Wk = (const float*)d_in[3];
  const float* bk = (const float*)d_in[4];
  const float* Wv = (const float*)d_in[5];
  const float* bv = (const float*)d_in[6];
  const float* Wo = (const float*)d_in[7];
  const float* bo = (const float*)d_in[8];
  float* out = (float*)d_out;

  const size_t MC = (size_t)M_DIM * C_DIM; // 8,388,608
  const size_t CC = (size_t)C_DIM * C_DIM; // 1,048,576

  f16* X16 = (f16*)d_ws;
  f16* WqT = X16 + MC;
  f16* WkT = WqT + CC;
  f16* WvT = WkT + CC;
  f16* WoT = WvT + CC;
  f16* Q16 = WoT + CC;  // head-major [B,H,P,64]
  f16* K16 = Q16 + MC;  // head-major [B,H,P,64]
  f16* V16 = K16 + MC;  // head-major transposed [B,H,64,P]
  f16* Y16 = V16 + MC;  // token-major [M,C]

  cvt_f32_f16<<<(int)(MC / 1024), 256, 0, stream>>>(x, X16, (int)MC);
  cvtT_f32_f16<<<(int)(CC / 256), 256, 0, stream>>>(Wq, WqT);
  cvtT_f32_f16<<<(int)(CC / 256), 256, 0, stream>>>(Wk, WkT);
  cvtT_f32_f16<<<(int)(CC / 256), 256, 0, stream>>>(Wv, WvT);
  cvtT_f32_f16<<<(int)(CC / 256), 256, 0, stream>>>(Wo, WoT);

  dim3 gg(C_DIM / 128, M_DIM / 128); // 8 x 64
  gemm128<1><<<gg, 256, 0, stream>>>(X16, WqT, bq, nullptr, Q16);
  gemm128<1><<<gg, 256, 0, stream>>>(X16, WkT, bk, nullptr, K16);
  gemm128<2><<<gg, 256, 0, stream>>>(X16, WvT, bv, nullptr, V16);

  attn64<<<dim3(P_DIM / 64, B_DIM * NHEADS), 128, 0, stream>>>(Q16, K16, V16, Y16);

  gemm128<0><<<gg, 256, 0, stream>>>(Y16, WoT, bo, out, nullptr);
}
